// ArsenalDecoderLayer_14396730376715
// MI455X (gfx1250) — matrix operand order repaired and checked by reading, not compiled, not run on hardware
//
#include <hip/hip_runtime.h>

typedef unsigned short u16;
typedef __attribute__((ext_vector_type(16))) __bf16 v16bf;
typedef __attribute__((ext_vector_type(8)))  float  v8f;

union Frag { uint4 q[2]; v16bf bf; };

// Async global->LDS copy path (gfx1250 GLOBAL_LOAD_ASYNC_TO_LDS_B128, ASYNCcnt).
// Guarded so compilation falls back to load+ds_store if the builtin is absent.
#if defined(__gfx1250__) && __has_builtin(__builtin_amdgcn_global_load_async_to_lds_b128) && __has_builtin(__builtin_amdgcn_s_wait_asynccnt)
#define ASYNC_COPY 1
#else
#define ASYNC_COPY 0
#endif

typedef int ivec4 __attribute__((vector_size(16)));

__device__ __forceinline__ void cp16(const u16* g, u16* l) {
#if ASYNC_COPY
    __builtin_amdgcn_global_load_async_to_lds_b128(
        (__attribute__((address_space(1))) ivec4*)g,
        (__attribute__((address_space(3))) ivec4*)l, 0, 0);
#else
    *(uint4*)l = *(const uint4*)g;
#endif
}

__device__ __forceinline__ void async_fence() {
#if ASYNC_COPY
    __builtin_amdgcn_s_wait_asynccnt(0);
#endif
}

__device__ __forceinline__ u16 f2bf(float f) {
    unsigned u = __float_as_uint(f);
    u += 0x7FFFu + ((u >> 16) & 1u);      // round-to-nearest-even
    return (u16)(u >> 16);
}

__device__ __forceinline__ float redmax16(float v) {
    #pragma unroll
    for (int o = 1; o < 16; o <<= 1) v = fmaxf(v, __shfl_xor(v, o, 32));
    return v;
}
__device__ __forceinline__ float redsum16(float v) {
    #pragma unroll
    for (int o = 1; o < 16; o <<= 1) v += __shfl_xor(v, o, 32);
    return v;
}

// ---------------------------------------------------------------------------
// Weight convert: fp32 W[K][N] -> bf16 Wt[N][K]
// ---------------------------------------------------------------------------
__global__ void wconv_kernel(const float* __restrict__ w, u16* __restrict__ wt,
                             int K, int N) {
    int i = blockIdx.x * 256 + threadIdx.x;
    if (i >= K * N) return;
    int k = i / N, n = i - k * N;
    wt[(size_t)n * K + k] = f2bf(w[i]);
}

// ---------------------------------------------------------------------------
// LayerNorm over rows of 1024, fp32 in -> bf16 out
// ---------------------------------------------------------------------------
__global__ void ln_kernel(const float* __restrict__ x, const float* __restrict__ g,
                          const float* __restrict__ sh, u16* __restrict__ out) {
    __shared__ float red[256];
    int row = blockIdx.x, t = threadIdx.x;
    const float* xr = x + (size_t)row * 1024;
    float v[4]; float s = 0.f;
    #pragma unroll
    for (int i = 0; i < 4; i++) { v[i] = xr[t + i * 256]; s += v[i]; }
    red[t] = s; __syncthreads();
    for (int o = 128; o > 0; o >>= 1) { if (t < o) red[t] += red[t + o]; __syncthreads(); }
    float mean = red[0] * (1.0f / 1024.0f);
    __syncthreads();
    float sq = 0.f;
    #pragma unroll
    for (int i = 0; i < 4; i++) { float d = v[i] - mean; sq += d * d; }
    red[t] = sq; __syncthreads();
    for (int o = 128; o > 0; o >>= 1) { if (t < o) red[t] += red[t + o]; __syncthreads(); }
    float rstd = rsqrtf(red[0] * (1.0f / 1024.0f) + 1e-5f);
    u16* orow = out + (size_t)row * 1024;
    #pragma unroll
    for (int i = 0; i < 4; i++) {
        int c = t + i * 256;
        orow[c] = f2bf((v[i] - mean) * rstd * g[c] + sh[c]);
    }
}

// ---------------------------------------------------------------------------
// WMMA bf16 GEMM: C[M,N] = act(scale*A@Bt^T + bias) + resid
// A bf16 [M][K], Bt bf16 [N][K].
// 256 threads = 8 waves (full WGP). Block tile 128x128, K-step 64,
// double-buffered LDS (64KB) fed by async global->LDS copies.
// Each wave computes 32x64 = 2x4 WMMA tiles (16 WMMAs per K-step).
// ---------------------------------------------------------------------------
__global__ void __launch_bounds__(256)
gemm_kernel(const u16* __restrict__ A, const u16* __restrict__ Bt,
            int M, int N, int K,
            const float* __restrict__ bias, const float* __restrict__ resid,
            float scale, int act,
            float* __restrict__ outF, u16* __restrict__ outB, int transV) {
    __shared__ u16 lA[2 * 128 * 64];
    __shared__ u16 lB[2 * 128 * 64];
    int m0 = blockIdx.y * 128, n0 = blockIdx.x * 128;
    int t = threadIdx.x, lane = t & 31, wave = t >> 5;
    int wm = (wave >> 1) * 32, wn = (wave & 1) * 64;
    int fm = lane & 15;
    int kbA = (lane < 16) ? 0 : 8, kbB = (lane < 16) ? 0 : 16, hi8 = (lane < 16) ? 0 : 8;
    v8f acc[2][4] = {};

    const u16* gA = A + (size_t)m0 * K;
    const u16* gB = Bt + (size_t)n0 * K;

    auto stage = [&](int buf, int k0) {
        u16* sa = &lA[buf * 128 * 64];
        u16* sb = &lB[buf * 128 * 64];
        #pragma unroll
        for (int i = 0; i < 4; i++) {
            int c = t + i * 256;                 // 1024 16B-chunks per matrix
            int row = c >> 3, off = (c & 7) * 8;
            cp16(gA + (size_t)row * K + k0 + off, sa + row * 64 + off);
            cp16(gB + (size_t)row * K + k0 + off, sb + row * 64 + off);
        }
    };

    auto compute = [&](int buf) {
        const u16* sa = &lA[buf * 128 * 64];
        const u16* sb = &lB[buf * 128 * 64];
        #pragma unroll
        for (int ks = 0; ks < 64; ks += 32) {
            Frag fa[2], fb[4];
            #pragma unroll
            for (int mi = 0; mi < 2; mi++) {
                const u16* p = sa + (wm + mi * 16 + fm) * 64 + ks;
                fa[mi].q[0] = *(const uint4*)(p + kbA);
                fa[mi].q[1] = *(const uint4*)(p + 16 + kbA);
            }
            #pragma unroll
            for (int ni = 0; ni < 4; ni++) {
                const u16* p = sb + (wn + ni * 16 + fm) * 64 + ks + kbA;
                fb[ni].q[0] = *(const uint4*)p;
                fb[ni].q[1] = *(const uint4*)(p + 16);
            }
            #pragma unroll
            for (int mi = 0; mi < 2; mi++)
                #pragma unroll
                for (int ni = 0; ni < 4; ni++)
                    acc[mi][ni] = __builtin_amdgcn_wmma_f32_16x16x32_bf16(
                        false, fa[mi].bf, false, fb[ni].bf, (short)0, acc[mi][ni], false, false);
        }
    };

    int nsteps = K >> 6;
    stage(0, 0);
    async_fence();
    __syncthreads();
    int buf = 0;
    for (int s = 0; s < nsteps; s++) {
        if (s + 1 < nsteps) stage(buf ^ 1, (s + 1) * 64);
        if (s + 2 < nsteps) {                     // L2 prefetch hint, 2 steps ahead
            int row = t >> 3, off = (t & 7) * 8;
            __builtin_prefetch(gA + (size_t)row * K + (s + 2) * 64 + off, 0, 0);
            __builtin_prefetch(gB + (size_t)row * K + (s + 2) * 64 + off, 0, 0);
        }
        compute(buf);
        async_fence();
        __syncthreads();
        buf ^= 1;
    }

    #pragma unroll
    for (int mi = 0; mi < 2; mi++)
        #pragma unroll
        for (int ni = 0; ni < 4; ni++)
            #pragma unroll
            for (int r = 0; r < 8; r++) {
                int row = m0 + wm + mi * 16 + r + hi8;
                int col = n0 + wn + ni * 16 + fm;
                float v = acc[mi][ni][r] * scale;
                if (bias)  v += bias[col];
                if (act)   v = 0.5f * v * (1.0f + erff(v * 0.70710678118654752f));
                if (resid) v += resid[(size_t)row * N + col];
                if (outF)  outF[(size_t)row * N + col] = v;
                if (outB) {
                    u16 bv = f2bf(v);
                    if (!transV) outB[(size_t)row * N + col] = bv;
                    else {
                        int bb = row >> 11, si = row & 2047, hh = col >> 6, dd = col & 63;
                        outB[((size_t)(bb * 16 + hh) * 64 + dd) * 2048 + si] = bv;
                    }
                }
            }
}

// ---------------------------------------------------------------------------
// Flash-style causal attention. One wave per (b,h, 16-query tile).
// q bf16 [4096][1024] (pre-scaled by 1/8), k bf16 [4096][1024],
// vT bf16 [(b*16+h)*64 + d][2048], out bf16 [4096][1024].
// ---------------------------------------------------------------------------
__global__ void __launch_bounds__(32)
attn_kernel(const u16* __restrict__ qb, const u16* __restrict__ kbuf,
            const u16* __restrict__ vT, u16* __restrict__ ob) {
    const int S = 2048, H = 1024;
    __shared__ u16 pl[16 * 32];
    int lane = threadIdx.x;
    int bh = blockIdx.y, b = bh >> 4, h = bh & 15;
    int q0 = blockIdx.x * 16;
    int fm = lane & 15;
    int kbA = (lane < 16) ? 0 : 8, kbB = (lane < 16) ? 0 : 16, hi8 = (lane < 16) ? 0 : 8;

    Frag fq[2];
    #pragma unroll
    for (int c = 0; c < 2; c++) {
        const u16* p = qb + (size_t)(b * S + q0 + fm) * H + h * 64 + c * 32;
        fq[c].q[0] = *(const uint4*)(p + kbA);
        fq[c].q[1] = *(const uint4*)(p + 16 + kbA);
    }
    v8f acc[4] = {};
    float rmax[8], rsum[8];
    #pragma unroll
    for (int r = 0; r < 8; r++) { rmax[r] = -1e30f; rsum[r] = 0.f; }

    int nkb = (q0 + 47) >> 5;                 // key chunks of 32 needed (causal)
    for (int kblk = 0; kblk < nkb; kblk++) {
        if (kblk + 1 < nkb) {                 // prefetch next K/V chunk
            __builtin_prefetch(kbuf + (size_t)(b * S + (kblk + 1) * 32 + fm) * H + h * 64, 0, 0);
            __builtin_prefetch(vT + (size_t)(bh * 64 + fm) * S + (kblk + 1) * 32, 0, 0);
        }
        v8f s[2] = {};
        #pragma unroll
        for (int nt = 0; nt < 2; nt++)
            #pragma unroll
            for (int c = 0; c < 2; c++) {
                const u16* p = kbuf + (size_t)(b * S + kblk * 32 + nt * 16 + fm) * H
                               + h * 64 + c * 32 + kbA;
                Frag fk; fk.q[0] = *(const uint4*)p; fk.q[1] = *(const uint4*)(p + 16);
                s[nt] = __builtin_amdgcn_wmma_f32_16x16x32_bf16(
                    false, fq[c].bf, false, fk.bf, (short)0, s[nt], false, false);
            }
        // causal mask
        #pragma unroll
        for (int nt = 0; nt < 2; nt++)
            #pragma unroll
            for (int r = 0; r < 8; r++) {
                int qi = q0 + r + hi8;
                int ki = kblk * 32 + nt * 16 + fm;
                if (ki > qi) s[nt][r] = -1e30f;
            }
        // online softmax stats (rows map as r + 8*(lane>=16); 16-lane reductions)
        #pragma unroll
        for (int r = 0; r < 8; r++) {
            float v = redmax16(fmaxf(s[0][r], s[1][r]));
            float nm = fmaxf(rmax[r], v);
            float sc = __expf(rmax[r] - nm);
            rmax[r] = nm;
            float p0 = __expf(s[0][r] - nm);
            float p1 = __expf(s[1][r] - nm);
            s[0][r] = p0; s[1][r] = p1;
            rsum[r] = rsum[r] * sc + redsum16(p0 + p1);
            #pragma unroll
            for (int tt = 0; tt < 4; tt++) acc[tt][r] *= sc;
        }
        __syncthreads();          // prior iteration's P-fragment reads done
        #pragma unroll
        for (int nt = 0; nt < 2; nt++)
            #pragma unroll
            for (int r = 0; r < 8; r++)
                pl[(r + hi8) * 32 + nt * 16 + fm] = f2bf(s[nt][r]);
        __syncthreads();
        Frag fp;
        { const u16* p = &pl[fm * 32];
          fp.q[0] = *(const uint4*)(p + kbA);
          fp.q[1] = *(const uint4*)(p + 16 + kbA); }
        #pragma unroll
        for (int tt = 0; tt < 4; tt++) {
            const u16* pv = vT + (size_t)(bh * 64 + tt * 16 + fm) * S + kblk * 32 + kbA;
            Frag fv; fv.q[0] = *(const uint4*)pv; fv.q[1] = *(const uint4*)(pv + 16);
            acc[tt] = __builtin_amdgcn_wmma_f32_16x16x32_bf16(
                false, fp.bf, false, fv.bf, (short)0, acc[tt], false, false);
        }
    }
    #pragma unroll
    for (int tt = 0; tt < 4; tt++)
        #pragma unroll
        for (int r = 0; r < 8; r++) {
            float v = acc[tt][r] / rsum[r];
            int row = q0 + r + hi8;
            int col = h * 64 + tt * 16 + fm;
            ob[(size_t)(b * S + row) * H + col] = f2bf(v);
        }
}

// ---------------------------------------------------------------------------
extern "C" void kernel_launch(void* const* d_in, const int* in_sizes, int n_in,
                              void* d_out, int out_size, void* d_ws, size_t ws_size,
                              hipStream_t stream) {
    const float* hidden = (const float*)d_in[0];
    const float* Wq  = (const float*)d_in[1];
    const float* Wk  = (const float*)d_in[2];
    const float* Wv  = (const float*)d_in[3];
    const float* Wo  = (const float*)d_in[4];
    const float* W1  = (const float*)d_in[5];
    const float* b1  = (const float*)d_in[6];
    const float* W2  = (const float*)d_in[7];
    const float* b2  = (const float*)d_in[8];
    const float* l1s = (const float*)d_in[9];
    const float* l1b = (const float*)d_in[10];
    const float* l2s = (const float*)d_in[11];
    const float* l2b = (const float*)d_in[12];
    (void)in_sizes; (void)n_in; (void)out_size; (void)ws_size;

    char* ws = (char*)d_ws; size_t off = 0;
    auto carve = [&](size_t bytes) -> char* {
        char* p = ws + off; off = (off + bytes + 255) & ~(size_t)255; return p;
    };
    u16*  wqT  = (u16*)carve(1024ull * 1024 * 2);
    u16*  wkT  = (u16*)carve(1024ull * 1024 * 2);
    u16*  wvT  = (u16*)carve(1024ull * 1024 * 2);
    u16*  woT  = (u16*)carve(1024ull * 1024 * 2);
    u16*  w1T  = (u16*)carve(1024ull * 4096 * 2);
    u16*  w2T  = (u16*)carve(4096ull * 1024 * 2);
    u16*  xbf  = (u16*)carve(4096ull * 1024 * 2);
    u16*  qbf  = (u16*)carve(4096ull * 1024 * 2);
    u16*  kbf  = (u16*)carve(4096ull * 1024 * 2);
    u16*  vTb  = (u16*)carve(4096ull * 1024 * 2);
    u16*  atb  = (u16*)carve(4096ull * 1024 * 2);
    float* hid2 = (float*)carve(4096ull * 1024 * 4);
    u16*  ybf  = (u16*)carve(4096ull * 1024 * 2);
    u16*  h1b  = (u16*)carve(4096ull * 4096 * 2);

    // weights -> bf16 transposed [N][K]
    wconv_kernel<<<4096, 256, 0, stream>>>(Wq, wqT, 1024, 1024);
    wconv_kernel<<<4096, 256, 0, stream>>>(Wk, wkT, 1024, 1024);
    wconv_kernel<<<4096, 256, 0, stream>>>(Wv, wvT, 1024, 1024);
    wconv_kernel<<<4096, 256, 0, stream>>>(Wo, woT, 1024, 1024);
    wconv_kernel<<<16384, 256, 0, stream>>>(W1, w1T, 1024, 4096);
    wconv_kernel<<<16384, 256, 0, stream>>>(W2, w2T, 4096, 1024);

    ln_kernel<<<4096, 256, 0, stream>>>(hidden, l1s, l1b, xbf);

    dim3 blk(256);
    // Q (pre-scaled by 1/sqrt(64)), K, V (stored head-transposed)
    gemm_kernel<<<dim3(8, 32), blk, 0, stream>>>(xbf, wqT, 4096, 1024, 1024,
        nullptr, nullptr, 0.125f, 0, nullptr, qbf, 0);
    gemm_kernel<<<dim3(8, 32), blk, 0, stream>>>(xbf, wkT, 4096, 1024, 1024,
        nullptr, nullptr, 1.0f, 0, nullptr, kbf, 0);
    gemm_kernel<<<dim3(8, 32), blk, 0, stream>>>(xbf, wvT, 4096, 1024, 1024,
        nullptr, nullptr, 1.0f, 0, nullptr, vTb, 1);

    attn_kernel<<<dim3(128, 32), 32, 0, stream>>>(qbf, kbf, vTb, atb);

    // O-projection + residual -> hid2 (fp32)
    gemm_kernel<<<dim3(8, 32), blk, 0, stream>>>(atb, woT, 4096, 1024, 1024,
        nullptr, hidden, 1.0f, 0, hid2, nullptr, 0);

    ln_kernel<<<4096, 256, 0, stream>>>(hid2, l2s, l2b, ybf);

    // FFN1: GELU(y@W1 + b1) -> bf16
    gemm_kernel<<<dim3(32, 32), blk, 0, stream>>>(ybf, w1T, 4096, 4096, 1024,
        b1, nullptr, 1.0f, 1, nullptr, h1b, 0);
    // FFN2: h1@W2 + b2 + hid2 -> fp32 output
    gemm_kernel<<<dim3(8, 32), blk, 0, stream>>>(h1b, w2T, 4096, 1024, 4096,
        b2, hid2, 1.0f, 0, (float*)d_out, nullptr, 0);
}
